// Experts_83502754169015
// MI455X (gfx1250) — matrix operand order repaired and checked by reading, not compiled, not run on hardware
//
#include <hip/hip_runtime.h>
#include <hip/hip_bf16.h>

typedef __attribute__((ext_vector_type(16))) __bf16 v16bf;
typedef __attribute__((ext_vector_type(8)))  float  v8f;

#define B_TOT 4096
#define D_DIM 128
#define H_DIM 64
#define N_EXP 64
#define G_DIM 256   // 4*H
#define BT    64    // batch rows per block

// Pack 8 contiguous f32 -> bf16 into fragment elements [off, off+8)
__device__ __forceinline__ void pack8(v16bf& r, int off, const float* __restrict__ p) {
  const float4* q = (const float4*)p;
  float4 v0 = q[0];
  float4 v1 = q[1];
  r[off + 0] = (__bf16)v0.x; r[off + 1] = (__bf16)v0.y;
  r[off + 2] = (__bf16)v0.z; r[off + 3] = (__bf16)v0.w;
  r[off + 4] = (__bf16)v1.x; r[off + 5] = (__bf16)v1.y;
  r[off + 6] = (__bf16)v1.z; r[off + 7] = (__bf16)v1.w;
}

__device__ __forceinline__ float sigf(float x) {
  return 1.0f / (1.0f + __expf(-x));
}

__global__ __launch_bounds__(256) void Experts_83502754169015_kernel(
    const float* __restrict__ x,   const float* __restrict__ h0,
    const float* __restrict__ c0,  const float* __restrict__ Wih,
    const float* __restrict__ Whh, const float* __restrict__ bih,
    const float* __restrict__ bhh, float* __restrict__ out)
{
  __shared__ float gsm[BT][G_DIM + 4];   // 64 x 260 f32 = 66.6 KB

  const int n     = blockIdx.x >> 6;     // expert
  const int btile = blockIdx.x & 63;     // 64-row batch tile
  const int tid   = threadIdx.x;
  const int wave  = tid >> 5;            // 0..7 (wave32)
  const int lane  = tid & 31;
  const int l16   = lane & 15;
  const bool hi   = lane >= 16;

  const int mt   = wave & 3;             // M-tile 0..3 within 64-row block
  const int ngrp = wave >> 2;            // gate-column tile group: 0 or 1
  const int rowA = btile * BT + mt * 16 + l16;   // global batch row this lane supplies

  // ---- A fragments: K=192 as six K=32 steps (4 from x, 2 from h0) ----
  // 16-bit A layout: lane<16 -> K {k0..k0+7, k0+16..k0+23}; lane>=16 -> +8
  v16bf afrag[6];
#pragma unroll
  for (int s = 0; s < 4; ++s) {
    const float* p = x + (size_t)rowA * D_DIM + s * 32 + (hi ? 8 : 0);
    pack8(afrag[s], 0, p);
    pack8(afrag[s], 8, p + 16);
  }
#pragma unroll
  for (int s = 0; s < 2; ++s) {
    const float* p = h0 + (size_t)rowA * H_DIM + s * 32 + (hi ? 8 : 0);
    pack8(afrag[4 + s], 0, p);
    pack8(afrag[4 + s], 8, p + 16);
  }

  // ---- GEMM: 8 gate-column tiles per wave, 6 WMMAs each (48 v_wmma/wave) ----
#pragma unroll
  for (int t = 0; t < 8; ++t) {
    const int nt = ngrp * 8 + t;         // gate-column tile 0..15
    const int g  = nt * 16 + l16;        // gate column this lane supplies for B
    // 16-bit B layout: lane<16 -> K {0..7, 16..23}; lane>=16 -> +8 (matches A),
    // column = lane%16 -> contiguous in W's row-major [g, k] storage.
    const float* wih_row = Wih + ((size_t)n * G_DIM + g) * D_DIM + (hi ? 8 : 0);
    const float* whh_row = Whh + ((size_t)n * G_DIM + g) * H_DIM + (hi ? 8 : 0);
    v8f acc = {};
#pragma unroll
    for (int s = 0; s < 4; ++s) {
      v16bf bfrag;
      pack8(bfrag, 0, wih_row + s * 32);
      pack8(bfrag, 8, wih_row + s * 32 + 16);
      acc = __builtin_amdgcn_wmma_f32_16x16x32_bf16(
          false, afrag[s], false, bfrag, (short)0, acc, false, false);
    }
#pragma unroll
    for (int s = 0; s < 2; ++s) {
      v16bf bfrag;
      pack8(bfrag, 0, whh_row + s * 32);
      pack8(bfrag, 8, whh_row + s * 32 + 16);
      acc = __builtin_amdgcn_wmma_f32_16x16x32_bf16(
          false, afrag[4 + s], false, bfrag, (short)0, acc, false, false);
    }
    // C/D layout: VGPR r, lane -> row (r + hi*8), col lane%16
#pragma unroll
    for (int r = 0; r < 8; ++r) {
      gsm[mt * 16 + r + (hi ? 8 : 0)][nt * 16 + l16] = acc[r];
    }
  }

  __syncthreads();

  // ---- LSTM element-wise epilogue: 64x64 h-values, 16 per thread ----
  const int h       = tid & 63;          // constant per thread (256 % 64 == 0)
  const int quarter = tid >> 6;
  const float bi = bih[n * G_DIM +       h] + bhh[n * G_DIM +       h];
  const float bf = bih[n * G_DIM +  64 + h] + bhh[n * G_DIM +  64 + h];
  const float bg = bih[n * G_DIM + 128 + h] + bhh[n * G_DIM + 128 + h];
  const float bo = bih[n * G_DIM + 192 + h] + bhh[n * G_DIM + 192 + h];

  const size_t BNH = (size_t)B_TOT * N_EXP * H_DIM;   // one output plane
#pragma unroll
  for (int e = 0; e < 16; ++e) {
    const int rl = e * 4 + quarter;      // local batch row 0..63
    const int b  = btile * BT + rl;
    const float gi = gsm[rl][h]        + bi;
    const float gf = gsm[rl][ 64 + h]  + bf;
    const float gg = gsm[rl][128 + h]  + bg;
    const float go = gsm[rl][192 + h]  + bo;
    const float c0v = c0[(size_t)b * H_DIM + h];
    const float cn = sigf(gf) * c0v + sigf(gi) * tanhf(gg);
    const float hn = sigf(go) * tanhf(cn);
    const size_t oi = (size_t)b * (N_EXP * H_DIM) + (size_t)n * H_DIM + h;
    out[oi]           = hn;   // output
    out[oi + BNH]     = hn;   // h (aliases output in reference)
    out[oi + 2 * BNH] = hn;   // c (aliases output in reference)
  }
}

extern "C" void kernel_launch(void* const* d_in, const int* in_sizes, int n_in,
                              void* d_out, int out_size, void* d_ws, size_t ws_size,
                              hipStream_t stream) {
  const float* x   = (const float*)d_in[0];
  const float* h0  = (const float*)d_in[1];
  const float* c0  = (const float*)d_in[2];
  const float* Wih = (const float*)d_in[3];
  const float* Whh = (const float*)d_in[4];
  const float* bih = (const float*)d_in[5];
  const float* bhh = (const float*)d_in[6];
  float* out = (float*)d_out;

  // one block per (expert, 64-row batch tile): 64 * 64 = 4096 blocks, 8 waves each
  Experts_83502754169015_kernel<<<dim3(N_EXP * (B_TOT / BT)), dim3(256), 0, stream>>>(
      x, h0, c0, Wih, Whh, bih, bhh, out);
}
